// ModulatedConv2d_42262478193471
// MI455X (gfx1250) — compile-verified, matrix operand order checked by reading, not run on hardware
//
#include <hip/hip_runtime.h>

#define B_      16
#define CIN_    256
#define COUT_   256
#define STYLE_  512
#define NK_     4
#define H_      64
#define W_      64
#define P_      (H_ * W_)     // 4096 pixels
#define KK_     (CIN_ * 9)    // 2304, tap-major: k = tap*256 + cin
#define EPS_    1e-8f

#define BM 128                // cout tile
#define BN 128                // pixel tile
#define BK 32                 // K chunk (one WMMA K)
#define PITCH 48              // LDS row pitch (elements): 96B, 16B-aligned rows
#define WP 68                 // padded input row pitch (floats)
#define HP 66                 // padded input rows
#define CHP (HP * WP)         // 4488 floats per padded channel plane
#define NIT 72                // K iterations

typedef __attribute__((ext_vector_type(16))) __bf16          v16bf;
typedef __attribute__((ext_vector_type(8)))  float           v8f;
typedef __attribute__((ext_vector_type(4)))  unsigned int    u4;

static __device__ inline unsigned short f2bf(float f) {
    union { __bf16 b; unsigned short u; } v;
    v.b = (__bf16)f;          // hardware RNE convert (v_cvt_pk_bf16_f32)
    return v.u;
}

// Async global->LDS byte copy (GLOBAL_LOAD_ASYNC_TO_LDS_B128, ASYNCcnt).
// VDST operand = per-lane LDS byte address = low 32 bits of the flat pointer.
static __device__ inline void async_copy_b128(const void* gptr, void* lptr) {
    unsigned lds            = (unsigned)(unsigned long long)lptr;
    unsigned long long ga   = (unsigned long long)gptr;
    asm volatile("global_load_async_to_lds_b128 %0, %1, off"
                 :: "v"(lds), "v"(ga) : "memory");
}
static __device__ inline void wait_async0() {
#if __has_builtin(__builtin_amdgcn_s_wait_asynccnt)
    __builtin_amdgcn_s_wait_asynccnt(0);
#else
    asm volatile("s_wait_asynccnt 0" ::: "memory");
#endif
}

// ---------------------------------------------------------------------------
// Phase 1: selection softmax [B,NK] and modulation vector s [B,CIN]
// ---------------------------------------------------------------------------
__global__ __launch_bounds__(256)
void prep_kernel(const float* __restrict__ style,
                 const float* __restrict__ affine_w,
                 const float* __restrict__ affine_b,
                 const float* __restrict__ mod_w,
                 const float* __restrict__ mod_b,
                 float* __restrict__ sel,
                 float* __restrict__ smod) {
    const int b = blockIdx.x;
    const int t = threadIdx.x;
    const float* st = style + b * STYLE_;

    __shared__ float red[NK_][256];
    float p[NK_];
#pragma unroll
    for (int n = 0; n < NK_; ++n) p[n] = 0.0f;
    for (int j = t; j < STYLE_; j += 256) {
        float sv = st[j];
#pragma unroll
        for (int n = 0; n < NK_; ++n) p[n] += sv * affine_w[n * STYLE_ + j];
    }
#pragma unroll
    for (int n = 0; n < NK_; ++n) red[n][t] = p[n];
    __syncthreads();
    for (int s = 128; s > 0; s >>= 1) {
        if (t < s) {
#pragma unroll
            for (int n = 0; n < NK_; ++n) red[n][t] += red[n][t + s];
        }
        __syncthreads();
    }
    if (t == 0) {
        float lg[NK_], mx = -1e30f;
#pragma unroll
        for (int n = 0; n < NK_; ++n) { lg[n] = red[n][0] + affine_b[n]; mx = fmaxf(mx, lg[n]); }
        float sum = 0.0f;
#pragma unroll
        for (int n = 0; n < NK_; ++n) { lg[n] = expf(lg[n] - mx); sum += lg[n]; }
        float inv = 1.0f / sum;
#pragma unroll
        for (int n = 0; n < NK_; ++n) sel[b * NK_ + n] = lg[n] * inv;
    }

    if (t < CIN_) {
        const float mod_scale = 0.04419417382415922f; // 1/sqrt(512)
        float acc = mod_b[t];
        const float* mw = mod_w + t * STYLE_;
        for (int j = 0; j < STYLE_; ++j) acc += st[j] * mw[j] * mod_scale;
        smod[b * CIN_ + t] = acc;
    }
}

// ---------------------------------------------------------------------------
// Phase 2: modulated + demodulated weights, bf16, tap-major layout
// Aw[b][o][tap*256 + cin].  One block = (b, o); thread i = input channel i.
// ---------------------------------------------------------------------------
__global__ __launch_bounds__(256)
void wmod_kernel(const float* __restrict__ weight,
                 const float* __restrict__ sel,
                 const float* __restrict__ smod,
                 unsigned short* __restrict__ Aw) {
    const int o = blockIdx.x;
    const int b = blockIdx.y;
    const int i = threadIdx.x;
    const float scale = 0.020833333333333332f;  // 1/sqrt(CIN*9) = 1/48
    const int nstride = COUT_ * CIN_ * 9;

    const float s0 = sel[b * NK_ + 0];
    const float s1 = sel[b * NK_ + 1];
    const float s2 = sel[b * NK_ + 2];
    const float s3 = sel[b * NK_ + 3];
    const float sv = smod[b * CIN_ + i];

    const float* w0 = weight + ((size_t)(o * CIN_ + i)) * 9;
    float wv[9];
    float sq = 0.0f;
#pragma unroll
    for (int kk = 0; kk < 9; ++kk) {
        float aw = s0 * w0[kk]
                 + s1 * w0[kk + nstride]
                 + s2 * w0[kk + 2 * nstride]
                 + s3 * w0[kk + 3 * nstride];
        float v = scale * aw * sv;
        wv[kk] = v;
        sq += v * v;
    }

    __shared__ float red[256];
    __shared__ float dsh;
    red[i] = sq;
    __syncthreads();
    for (int s = 128; s > 0; s >>= 1) {
        if (i < s) red[i] += red[i + s];
        __syncthreads();
    }
    if (i == 0) dsh = rsqrtf(red[0] + EPS_);
    __syncthreads();
    const float d = dsh;

    unsigned short* dst = Aw + ((size_t)(b * COUT_ + o)) * KK_ + i;  // tap-major
#pragma unroll
    for (int kk = 0; kk < 9; ++kk) dst[kk * CIN_] = f2bf(wv[kk] * d);
}

// ---------------------------------------------------------------------------
// Phase 3: zero-pad input into [B][CIN][66][68] fp32
// ---------------------------------------------------------------------------
__global__ __launch_bounds__(256)
void pad_kernel(const float* __restrict__ in, float* __restrict__ Xpad) {
    const int bc = blockIdx.x;                 // b*CIN + c
    const float* src = in + (size_t)bc * P_;
    float* dst = Xpad + (size_t)bc * CHP;
    for (int e = threadIdx.x; e < CHP; e += 256) dst[e] = 0.0f;
    __syncthreads();
    for (int e = threadIdx.x; e < P_; e += 256) {
        int yy = e >> 6, xx = e & 63;
        dst[(yy + 1) * WP + (xx + 1)] = src[e];
    }
}

// ---------------------------------------------------------------------------
// Phase 4: implicit-GEMM conv.  Block 128x128, 8 waves (4Mx2N), wave = 32x64
// Double-buffered LDS, async global->LDS weight staging, 1 barrier/K-step,
// 8 v_wmma_f32_16x16x32_bf16 per wave per K-step, 72 K-steps.
// ---------------------------------------------------------------------------
__global__ __launch_bounds__(256)
void conv_kernel(const float* __restrict__ Xpad,
                 const unsigned short* __restrict__ Aw,
                 float* __restrict__ out) {
    const int b  = blockIdx.z;
    const int o0 = blockIdx.y * BM;
    const int p0 = blockIdx.x * BN;
    const int t  = threadIdx.x;
    const int lane = t & 31;
    const int wave = t >> 5;
    const int wm   = wave & 3;     // 4 M-groups of 32 rows
    const int wn   = wave >> 2;    // 2 N-groups of 64 cols
    const int half = lane >> 4;    // ISA 16-bit fragment lane group
    const int mn   = lane & 15;    // M (A) / N (B) within 16-tile

    __shared__ __align__(16) unsigned short Atile[2][BM * PITCH];
    __shared__ __align__(16) unsigned short Btile[2][BN * PITCH];

    // A staging: thread -> (row, 16-col half), 2x async b128 copies
    const int ar = t >> 1;              // 0..127
    const int ac = (t & 1) * 16;        // 0 or 16
    const unsigned short* aptr = Aw + ((size_t)(b * COUT_ + o0 + ar)) * KK_ + ac;

    // B staging: thread -> fixed pixel, 16 consecutive channels of the chunk
    const int pl  = t & 127;            // pixel in tile
    const int klb = (t >> 7) * 16;      // 0 or 16 within K-chunk
    const int p = p0 + pl;
    const int y = p >> 6;
    const int x = p & 63;
    const float* xbase = Xpad + (size_t)b * CIN_ * CHP + (y + 1) * WP + (x + 1);

    // ---- pipeline helpers ----
    auto stageA = [&](int it, int buf) {
        const unsigned short* g = aptr + it * BK;
        unsigned short* l = &Atile[buf][ar * PITCH + ac];
        async_copy_b128(g,     l);
        async_copy_b128(g + 8, l + 8);
    };
    auto loadB = [&](int it, unsigned short* bs) {
        const int kk = it >> 3;
        const int ch = it & 7;
        const int dy = kk / 3 - 1;
        const int dx = kk - (kk / 3) * 3 - 1;
        const float* xp = xbase + (size_t)(ch * 32 + klb) * CHP + dy * WP + dx;
#pragma unroll
        for (int j = 0; j < 16; ++j)
            bs[j] = f2bf(xp[(size_t)j * CHP]);     // coalesced across lanes
    };
    auto storeB = [&](int buf, const unsigned short* bs) {
        union { unsigned short s[16]; u4 v[2]; } bp;
#pragma unroll
        for (int j = 0; j < 16; ++j) bp.s[j] = bs[j];
        u4* bd = (u4*)&Btile[buf][pl * PITCH + klb];
        bd[0] = bp.v[0]; bd[1] = bp.v[1];
    };

    v8f acc[2][4] = {};

    // ---- prologue: stage chunk 0 into buffer 0 ----
    {
        unsigned short bs[16];
        stageA(0, 0);
        loadB(0, bs);
        storeB(0, bs);
        wait_async0();
        __syncthreads();
    }

    for (int it = 0; it < NIT; ++it) {
        const int cur = it & 1;
        const int nxt = cur ^ 1;

        // ---- prefetch next chunk (overlaps with WMMA below) ----
        unsigned short bn[16];
        if (it < NIT - 1) {
            stageA(it + 1, nxt);       // async, lands directly in LDS
            loadB(it + 1, bn);         // fp32 loads + cvt in VGPRs
        }

        // ---- fragments from current buffer ----
        // lanes 0-15: K = {0..7, 16..23}; lanes 16-31: K = {8..15, 24..31}
        union FR { u4 u[2]; v16bf v; };
        FR fa[2], fb[4];
#pragma unroll
        for (int mi = 0; mi < 2; ++mi) {
            const unsigned short* r = &Atile[cur][(wm * 32 + mi * 16 + mn) * PITCH];
            fa[mi].u[0] = *(const u4*)(r + half * 8);
            fa[mi].u[1] = *(const u4*)(r + 16 + half * 8);
        }
#pragma unroll
        for (int ni = 0; ni < 4; ++ni) {
            const unsigned short* r = &Btile[cur][(wn * 64 + ni * 16 + mn) * PITCH];
            fb[ni].u[0] = *(const u4*)(r + half * 8);
            fb[ni].u[1] = *(const u4*)(r + 16 + half * 8);
        }

#pragma unroll
        for (int mi = 0; mi < 2; ++mi)
#pragma unroll
            for (int ni = 0; ni < 4; ++ni)
                acc[mi][ni] = __builtin_amdgcn_wmma_f32_16x16x32_bf16(
                    false, fa[mi].v, false, fb[ni].v,
                    (short)0, acc[mi][ni], false, false);

        // ---- finish staging next buffer, one barrier per K-step ----
        if (it < NIT - 1) storeB(nxt, bn);
        wait_async0();
        __syncthreads();
    }

    // ---- epilogue: C/D layout VGPR r -> M = r + 8*half, N = mn ----
#pragma unroll
    for (int mi = 0; mi < 2; ++mi) {
        float* op = out + ((size_t)(b * COUT_ + o0 + wm * 32 + mi * 16)) * P_
                        + p0 + wn * 64 + mn;
#pragma unroll
        for (int r = 0; r < 8; ++r) {
#pragma unroll
            for (int ni = 0; ni < 4; ++ni)
                op[(size_t)(r + half * 8) * P_ + ni * 16] = acc[mi][ni][r];
        }
    }
}

extern "C" void kernel_launch(void* const* d_in, const int* in_sizes, int n_in,
                              void* d_out, int out_size, void* d_ws, size_t ws_size,
                              hipStream_t stream) {
    const float* input    = (const float*)d_in[0];
    const float* style    = (const float*)d_in[1];
    const float* weight   = (const float*)d_in[2];
    const float* affine_w = (const float*)d_in[3];
    const float* affine_b = (const float*)d_in[4];
    const float* mod_w    = (const float*)d_in[5];
    const float* mod_b    = (const float*)d_in[6];
    float* out = (float*)d_out;

    char* ws = (char*)d_ws;
    float* sel         = (float*)ws;                      // 64 floats
    float* smod        = (float*)(ws + 1024);             // 4096 floats
    unsigned short* Aw = (unsigned short*)(ws + 32768);   // 16*256*2304 bf16 (~18.9 MB)
    float* Xpad        = (float*)(ws + 32768 + (size_t)B_ * COUT_ * KK_ * 2); // ~73.5 MB

    prep_kernel<<<dim3(B_), dim3(256), 0, stream>>>(style, affine_w, affine_b,
                                                    mod_w, mod_b, sel, smod);
    wmod_kernel<<<dim3(COUT_, B_), dim3(256), 0, stream>>>(weight, sel, smod, Aw);
    pad_kernel<<<dim3(B_ * CIN_), dim3(256), 0, stream>>>(input, Xpad);
    conv_kernel<<<dim3(P_ / BN, COUT_ / BM, B_), dim3(256), 0, stream>>>(Xpad, Aw, out);
}
